// BertSelfAttention_45337674777147
// MI455X (gfx1250) — compile-verified, matrix operand order checked by reading, not run on hardware
//
#include <hip/hip_runtime.h>

#define S_LEN 4096
#define HID   768
#define NH    12
#define HD    64

typedef __attribute__((ext_vector_type(16))) __bf16   v16bf;
typedef __attribute__((ext_vector_type(8)))  float    v8f;
typedef __attribute__((ext_vector_type(4)))  float    v4f;
typedef __attribute__((ext_vector_type(4)))  unsigned v4u;

// 16-bit A/B fragment = two contiguous 8-element (16B) runs:
//   K = 8*(lane>>4) + 0..7   and   16 + 8*(lane>>4) + 0..7
union Frag { v16bf bf; v4u q[2]; };

__device__ __forceinline__ v16bf ld_frag(const __bf16* p0, const __bf16* p1) {
  Frag f;
  f.q[0] = *(const v4u*)p0;   // global_load_b128
  f.q[1] = *(const v4u*)p1;   // global_load_b128
  return f.bf;
}

#define WMMA_BF16(A, B, C) \
  __builtin_amdgcn_wmma_f32_16x16x32_bf16(false, (A), false, (B), (short)0, (C), false, false)

// ---------------------------------------------------------------------------
// Pre-pass A: hidden f32 -> bf16, row-major [s][k] (contraction dim innermost)
// ---------------------------------------------------------------------------
__global__ __launch_bounds__(256) void cvt_h_kernel(const float* __restrict__ H,
                                                    __bf16* __restrict__ Hbf) {
  int i = (blockIdx.x * 256 + threadIdx.x) * 4;
  v4f v = *(const v4f*)(H + i);
  __bf16* o = Hbf + i;
  o[0] = (__bf16)v.x; o[1] = (__bf16)v.y; o[2] = (__bf16)v.z; o[3] = (__bf16)v.w;
}

// ---------------------------------------------------------------------------
// Pre-pass B: W[k][o] f32 -> WT[o][k] bf16, LDS-tiled 32x32 transpose.
// blockIdx.z selects Q/K/V.
// ---------------------------------------------------------------------------
__global__ __launch_bounds__(256) void xpose_w_kernel(
    const float* __restrict__ Wq, const float* __restrict__ Wk,
    const float* __restrict__ Wv, __bf16* __restrict__ WT) {
  __shared__ float t[32][33];
  const float* W = (blockIdx.z == 0) ? Wq : (blockIdx.z == 1) ? Wk : Wv;
  __bf16* WTm = WT + (size_t)blockIdx.z * HID * HID;
  int tx = threadIdx.x & 31, ty = threadIdx.x >> 5;     // 32 x 8
  int k0 = blockIdx.x * 32, o0 = blockIdx.y * 32;
#pragma unroll
  for (int j = 0; j < 4; ++j)
    t[ty + j * 8][tx] = W[(k0 + ty + j * 8) * HID + o0 + tx];   // coalesced in o
  __syncthreads();
#pragma unroll
  for (int j = 0; j < 4; ++j)
    WTm[(o0 + ty + j * 8) * HID + k0 + tx] = (__bf16)t[tx][ty + j * 8]; // coalesced in k
}

// ---------------------------------------------------------------------------
// Kernel 1: fused QKV projection. One wave = one 16x16 output tile; all
// fragments are two b128 loads (bf16, contraction innermost). Q pre-scaled by
// 1/sqrt(HD). Q,K stored [h][s][d]; V stored transposed [h][d][s] so the
// attention PV B-fragments are contiguous too.
// ---------------------------------------------------------------------------
__global__ __launch_bounds__(256) void qkv_proj_kernel(
    const __bf16* __restrict__ Hbf, const __bf16* __restrict__ WT,
    const float* __restrict__ bq, const float* __restrict__ bk,
    const float* __restrict__ bv,
    __bf16* __restrict__ qws, __bf16* __restrict__ kws, __bf16* __restrict__ vwsT)
{
  const int lane  = threadIdx.x & 31;
  const int wave  = threadIdx.x >> 5;
  const int row0  = blockIdx.x * 16;
  const int strip = blockIdx.y;                  // 0..17, 6 strips per matrix
  const int mat   = strip / 6;                   // 0=Q 1=K 2=V
  const int col0  = (strip % 6) * 128 + wave * 16;
  const int khalf = lane >> 4;
  const int nn    = lane & 15;
  const int co    = khalf << 3;                  // chunk offset within 32-K step

  const float*  bias = (mat == 0) ? bq : (mat == 1) ? bk : bv;
  const __bf16* arow = Hbf + (size_t)(row0 + nn) * HID;
  const __bf16* brow = WT + (size_t)mat * HID * HID + (size_t)(col0 + nn) * HID;

  v8f acc = {};
#pragma unroll 4
  for (int kb = 0; kb < HID; kb += 32) {
    v16bf a = ld_frag(arow + kb + co, arow + kb + 16 + co);
    v16bf b = ld_frag(brow + kb + co, brow + kb + 16 + co);
    acc = WMMA_BF16(a, b, acc);
  }

  const float bv_  = bias[col0 + nn];
  const float scal = (mat == 0) ? 0.125f : 1.0f;      // fold 1/sqrt(64) into Q
#pragma unroll
  for (int v = 0; v < 8; ++v) {
    int s = row0 + v + co;                            // row m = v + 8*khalf
    int o = col0 + nn;
    int h = o >> 6, d = o & 63;
    float val = (acc[v] + bv_) * scal;
    if (mat == 2)
      vwsT[((size_t)h * HD + d) * S_LEN + s] = (__bf16)val;
    else
      ((mat == 0) ? qws : kws)[((size_t)h * S_LEN + s) * HD + d] = (__bf16)val;
  }
}

// ---------------------------------------------------------------------------
// Kernel 2: flash attention, 64-key blocks. One wave = 16 query rows of one
// head. Per block: 8 WMMA for S=Q*K^T (4 key-tiles x 2 k-steps), one online
// softmax update (shfl trees / rescale amortized over 64 keys), P transposed
// through wave-private LDS (b128 reads), 8 WMMA for O+=P*V.
// ---------------------------------------------------------------------------
__global__ __launch_bounds__(256) void attn_kernel(
    const __bf16* __restrict__ qws, const __bf16* __restrict__ kws,
    const __bf16* __restrict__ vwsT, const float* __restrict__ mask,
    float* __restrict__ out)
{
  __shared__ __align__(16) __bf16 Pt[8][16][64];   // per-wave 16x64, 128B rows

  const int lane  = threadIdx.x & 31;
  const int wave  = threadIdx.x >> 5;
  const int h     = blockIdx.y;
  const int q0    = blockIdx.x * 128 + wave * 16;
  const int khalf = lane >> 4;
  const int nn    = lane & 15;
  const int co    = khalf << 3;

  const __bf16* Qh  = qws  + (size_t)h * S_LEN * HD;
  const __bf16* Kh  = kws  + (size_t)h * S_LEN * HD;
  const __bf16* VhT = vwsT + (size_t)h * HD * S_LEN;

  // Q A-fragments, loaded once (d innermost -> two b128 each)
  const __bf16* qrow = Qh + (size_t)(q0 + nn) * HD;
  v16bf qa0 = ld_frag(qrow + co,      qrow + 16 + co);
  v16bf qa1 = ld_frag(qrow + 32 + co, qrow + 48 + co);

  v8f oa[4];
#pragma unroll
  for (int t = 0; t < 4; ++t) oa[t] = (v8f){};
  float m8[8], l8[8];
#pragma unroll
  for (int v = 0; v < 8; ++v) { m8[v] = -1e30f; l8[v] = 0.0f; }

  for (int kb = 0; kb < S_LEN; kb += 64) {
    if (kb + 64 < S_LEN) {
      __builtin_prefetch(&Kh[(size_t)(kb + 64 + nn) * HD], 0, 1);
      __builtin_prefetch(&VhT[(size_t)nn * S_LEN + kb + 64], 0, 1);
    }

    // ---- S = Q * K^T : 4 key-tiles x 2 k-steps = 8 WMMA
    v8f st[4];
#pragma unroll
    for (int t = 0; t < 4; ++t) {
      const __bf16* kr = Kh + (size_t)(kb + t * 16 + nn) * HD;
      v8f s = {};
      s = WMMA_BF16(qa0, ld_frag(kr + co,      kr + 16 + co), s);
      s = WMMA_BF16(qa1, ld_frag(kr + 32 + co, kr + 48 + co), s);
      st[t] = s;
    }

    float mk[4];
#pragma unroll
    for (int t = 0; t < 4; ++t) mk[t] = mask[kb + t * 16 + nn];

    // ---- online softmax (rows in VGPR index, key-cols in lanes)
#pragma unroll
    for (int v = 0; v < 8; ++v) {
      float a0 = st[0][v] + mk[0];
      float a1 = st[1][v] + mk[1];
      float a2 = st[2][v] + mk[2];
      float a3 = st[3][v] + mk[3];
      float mx = fmaxf(fmaxf(a0, a1), fmaxf(a2, a3));
#pragma unroll
      for (int x = 8; x >= 1; x >>= 1)        // xor<16 stays within half-wave
        mx = fmaxf(mx, __shfl_xor(mx, x, 32));
      float mnew = fmaxf(m8[v], mx);
      float sc   = __expf(m8[v] - mnew);
      float p0   = __expf(a0 - mnew);
      float p1   = __expf(a1 - mnew);
      float p2   = __expf(a2 - mnew);
      float p3   = __expf(a3 - mnew);
      st[0][v] = p0; st[1][v] = p1; st[2][v] = p2; st[3][v] = p3;
      float rs = (p0 + p1) + (p2 + p3);
#pragma unroll
      for (int x = 8; x >= 1; x >>= 1)
        rs += __shfl_xor(rs, x, 32);
      l8[v] = l8[v] * sc + rs;
      m8[v] = mnew;
#pragma unroll
      for (int t = 0; t < 4; ++t) oa[t][v] *= sc;
    }

    // ---- transpose P (C-layout -> A-layout) via wave-private LDS
#pragma unroll
    for (int v = 0; v < 8; ++v) {
      int m = v + co;
#pragma unroll
      for (int t = 0; t < 4; ++t)
        Pt[wave][m][t * 16 + nn] = (__bf16)st[t][v];
    }
    Frag pf0, pf1;                             // four ds_load_b128, same wave
    pf0.q[0] = *(const v4u*)&Pt[wave][nn][co];
    pf0.q[1] = *(const v4u*)&Pt[wave][nn][16 + co];
    pf1.q[0] = *(const v4u*)&Pt[wave][nn][32 + co];
    pf1.q[1] = *(const v4u*)&Pt[wave][nn][48 + co];

    // ---- O += P * V : 4 d-tiles x 2 k-steps = 8 WMMA (keys innermost in vT)
#pragma unroll
    for (int td = 0; td < 4; ++td) {
      const __bf16* vrow = VhT + (size_t)(td * 16 + nn) * S_LEN + kb;
      oa[td] = WMMA_BF16(pf0.bf, ld_frag(vrow + co,      vrow + 16 + co), oa[td]);
      oa[td] = WMMA_BF16(pf1.bf, ld_frag(vrow + 32 + co, vrow + 48 + co), oa[td]);
    }
  }

  // ---- epilogue: O / l, scatter to [s][h*64+d] f32
#pragma unroll
  for (int v = 0; v < 8; ++v) {
    int s = q0 + v + co;
    float inv = 1.0f / l8[v];
#pragma unroll
    for (int td = 0; td < 4; ++td)
      out[(size_t)s * HID + h * HD + td * 16 + nn] = oa[td][v] * inv;
  }
}

// ---------------------------------------------------------------------------
extern "C" void kernel_launch(void* const* d_in, const int* in_sizes, int n_in,
                              void* d_out, int out_size, void* d_ws, size_t ws_size,
                              hipStream_t stream) {
  (void)in_sizes; (void)n_in; (void)out_size; (void)ws_size;

  const float* H    = (const float*)d_in[0];
  const float* mask = (const float*)d_in[1];
  const float* Wq   = (const float*)d_in[2];
  const float* bq   = (const float*)d_in[3];
  const float* Wk   = (const float*)d_in[4];
  const float* bk   = (const float*)d_in[5];
  const float* Wv   = (const float*)d_in[6];
  const float* bv   = (const float*)d_in[7];
  float* out = (float*)d_out;

  const size_t per = (size_t)NH * S_LEN * HD;       // elements per head tensor
  __bf16* qws  = (__bf16*)d_ws;                     // [h][s][d]
  __bf16* kws  = qws  + per;                        // [h][s][d]
  __bf16* vwsT = kws  + per;                        // [h][d][s]
  __bf16* Hbf  = vwsT + per;                        // [s][k]
  __bf16* WT   = Hbf  + (size_t)S_LEN * HID;        // 3 x [o][k]

  cvt_h_kernel<<<(S_LEN * HID / 4) / 256, 256, 0, stream>>>(H, Hbf);
  xpose_w_kernel<<<dim3(HID / 32, HID / 32, 3), 256, 0, stream>>>(Wq, Wk, Wv, WT);

  dim3 g1(S_LEN / 16, (3 * HID) / 128);             // (256, 18)
  qkv_proj_kernel<<<g1, 256, 0, stream>>>(Hbf, WT, bq, bk, bv, qws, kws, vwsT);

  dim3 g2(S_LEN / 128, NH);                         // (32, 12)
  attn_kernel<<<g2, 256, 0, stream>>>(qws, kws, vwsT, mask, out);
}
